// ConditionalMamba1D_1589137900240
// MI455X (gfx1250) — matrix operand order repaired and checked by reading, not compiled, not run on hardware
//
#include <hip/hip_runtime.h>
#include <hip/hip_bf16.h>

typedef __attribute__((ext_vector_type(16))) _Float16 v16h;
typedef __attribute__((ext_vector_type(8)))  _Float16 v8h;
typedef __attribute__((ext_vector_type(8)))  float    v8f;

namespace {
constexpr int Bc   = 2;
constexpr int Lc   = 2048;
constexpr int Ec   = 1024;
constexpr int DIc  = 2048;
constexpr int DSc  = 16;
constexpr int DTRc = 64;
constexpr int Mc   = Bc * Lc;            // 4096 token rows
constexpr int XDBLc = DTRc + 2 * DSc;    // 96
}

// ---------------- f32 -> f16 conversion ----------------
__global__ __launch_bounds__(256) void cvt_f32_f16(const float* __restrict__ src,
                                                   _Float16* __restrict__ dst, int n) {
  int i = blockIdx.x * 256 + threadIdx.x;
  if (i < n) dst[i] = (_Float16)src[i];
}

// ---------------- FiLM params: g = desc@gamma_w.T + gamma_b, bt likewise ----------------
__global__ __launch_bounds__(256) void film_params(const float* __restrict__ desc,
                                                   const float* __restrict__ gw,
                                                   const float* __restrict__ gb,
                                                   const float* __restrict__ bw,
                                                   const float* __restrict__ bb,
                                                   float* __restrict__ g,
                                                   float* __restrict__ bt) {
  int idx = blockIdx.x * 256 + threadIdx.x;        // over Bc*Ec
  if (idx >= Bc * Ec) return;
  int b = idx / Ec, e = idx % Ec;
  const float* dr  = desc + (size_t)b * Ec;
  const float* gwr = gw   + (size_t)e * Ec;
  const float* bwr = bw   + (size_t)e * Ec;
  float gs = gb[e], bs = bb[e];
  for (int k = 0; k < Ec; ++k) { float dv = dr[k]; gs += dv * gwr[k]; bs += dv * bwr[k]; }
  g[idx] = gs; bt[idx] = bs;
}

// ---------------- LayerNorm + FiLM, emit f16 activations ----------------
__global__ __launch_bounds__(256) void ln_film(const float* __restrict__ seq,
                                               const float* __restrict__ ln_w,
                                               const float* __restrict__ ln_b,
                                               const float* __restrict__ g,
                                               const float* __restrict__ bt,
                                               _Float16* __restrict__ s16) {
  const int row = blockIdx.x;                      // b*Lc + l
  const int b = row / Lc;
  const int tid = threadIdx.x;
  const float* xr = seq + (size_t)row * Ec;
  float sum = 0.f, sumsq = 0.f;
  for (int e = tid; e < Ec; e += 256) { float v = xr[e]; sum += v; sumsq += v * v; }
  __shared__ float s1[256], s2[256];
  s1[tid] = sum; s2[tid] = sumsq;
  __syncthreads();
  for (int st = 128; st > 0; st >>= 1) {
    if (tid < st) { s1[tid] += s1[tid + st]; s2[tid] += s2[tid + st]; }
    __syncthreads();
  }
  const float mu   = s1[0] * (1.f / Ec);
  const float var  = s2[0] * (1.f / Ec) - mu * mu;
  const float rstd = rsqrtf(var + 1e-5f);
  const float* gr  = g  + (size_t)b * Ec;
  const float* btr = bt + (size_t)b * Ec;
  _Float16* out = s16 + (size_t)row * Ec;
  for (int e = tid; e < Ec; e += 256) {
    float v = (xr[e] - mu) * rstd * ln_w[e] + ln_b[e];
    v = gr[e] * v + btr[e];
    out[e] = (_Float16)v;
  }
}

// ---------------- WMMA GEMM: C(M,N) f32 = A(M,K) f16 * W(N,K)^T f16 ----------------
// mode 0: plain store.  mode 1: C = softplus(acc + bias[n])   (dt_proj epilogue)
__global__ __launch_bounds__(256) void wmma_gemm_nt(const _Float16* __restrict__ A,
                                                    const _Float16* __restrict__ W,
                                                    float* __restrict__ C,
                                                    int M, int N, int K,
                                                    const float* __restrict__ bias,
                                                    int mode) {
  const int lane = threadIdx.x & 31;
  const int wave = blockIdx.x * 8 + (threadIdx.x >> 5);
  const int tiles_n = N >> 5, tiles_m = M >> 5;
  if (wave >= tiles_m * tiles_n) return;           // uniform per wave: EXEC stays full
  const int tm = wave / tiles_n, tn = wave % tiles_n;
  const int m0 = tm << 5, n0 = tn << 5;
  const int h  = lane >> 4;                        // half-group (ISA A/B frag layout)
  const int ml = lane & 15;

  v8f acc00 = {}, acc01 = {}, acc10 = {}, acc11 = {};
  const _Float16* arow0 = A + (size_t)(m0 + ml) * K;
  const _Float16* arow1 = A + (size_t)(m0 + 16 + ml) * K;
  const _Float16* wrow0 = W + (size_t)(n0 + ml) * K + h * 8;
  const _Float16* wrow1 = W + (size_t)(n0 + 16 + ml) * K + h * 8;

  for (int k0 = 0; k0 < K; k0 += 32) {
    v16h a0, a1, b0, b1;
    {
      v8h lo = *(const v8h*)(arow0 + k0 + h * 8);
      v8h hi = *(const v8h*)(arow0 + k0 + 16 + h * 8);
#pragma unroll
      for (int i = 0; i < 8; ++i) { a0[i] = lo[i]; a0[8 + i] = hi[i]; }
    }
    {
      v8h lo = *(const v8h*)(arow1 + k0 + h * 8);
      v8h hi = *(const v8h*)(arow1 + k0 + 16 + h * 8);
#pragma unroll
      for (int i = 0; i < 8; ++i) { a1[i] = lo[i]; a1[8 + i] = hi[i]; }
    }
    {
      v8h lo = *(const v8h*)(wrow0 + k0);
      v8h hi = *(const v8h*)(wrow0 + k0 + 16);
#pragma unroll
      for (int i = 0; i < 8; ++i) { b0[i] = lo[i]; b0[8 + i] = hi[i]; }
    }
    {
      v8h lo = *(const v8h*)(wrow1 + k0);
      v8h hi = *(const v8h*)(wrow1 + k0 + 16);
#pragma unroll
      for (int i = 0; i < 8; ++i) { b1[i] = lo[i]; b1[8 + i] = hi[i]; }
    }
    acc00 = __builtin_amdgcn_wmma_f32_16x16x32_f16(false, a0, false, b0, (short)0, acc00, false, false);
    acc01 = __builtin_amdgcn_wmma_f32_16x16x32_f16(false, a0, false, b1, (short)0, acc01, false, false);
    acc10 = __builtin_amdgcn_wmma_f32_16x16x32_f16(false, a1, false, b0, (short)0, acc10, false, false);
    acc11 = __builtin_amdgcn_wmma_f32_16x16x32_f16(false, a1, false, b1, (short)0, acc11, false, false);
  }

  auto store_tile = [&](const v8f& acc, int mbase, int nbase) {
    const int n = nbase + ml;
#pragma unroll
    for (int r = 0; r < 8; ++r) {
      const int m = mbase + h * 8 + r;
      float v = acc[r];
      if (mode == 1) { v += bias[n]; v = (v > 20.f) ? v : log1pf(__expf(v)); }
      C[(size_t)m * N + n] = v;
    }
  };
  store_tile(acc00, m0,      n0);
  store_tile(acc01, m0,      n0 + 16);
  store_tile(acc10, m0 + 16, n0);
  store_tile(acc11, m0 + 16, n0 + 16);
}

// ---------------- causal depthwise conv (DC=4) + SiLU ----------------
__global__ __launch_bounds__(256) void conv_silu(const float* __restrict__ xz,
                                                 const float* __restrict__ cw,
                                                 const float* __restrict__ cb,
                                                 float* __restrict__ xc32,
                                                 _Float16* __restrict__ xc16) {
  int idx = blockIdx.x * 256 + threadIdx.x;        // over Bc*Lc*DIc, layout (b,l,d)
  if (idx >= Bc * Lc * DIc) return;
  const int d = idx % DIc;
  const int l = (idx / DIc) % Lc;
  const int b = idx / (DIc * Lc);
  float acc = cb[d];
  const float* w = cw + (size_t)d * 4;
#pragma unroll
  for (int j = 0; j < 4; ++j) {
    const int ll = l - 3 + j;
    if (ll >= 0) acc += w[j] * xz[((size_t)(b * Lc + ll) * 2 * DIc) + d];
  }
  const float s = acc * (1.f / (1.f + __expf(-acc)));
  xc32[idx] = s;
  xc16[idx] = (_Float16)s;
}

// ---------------- slice dt columns (0..63) of x_dbl into f16 ----------------
__global__ __launch_bounds__(256) void dt_slice(const float* __restrict__ xdbl,
                                                _Float16* __restrict__ dt16) {
  int idx = blockIdx.x * 256 + threadIdx.x;        // Mc * 64
  if (idx >= Mc * DTRc) return;
  const int r = idx >> 6, c = idx & 63;
  dt16[idx] = (_Float16)xdbl[(size_t)r * XDBLc + c];
}

// ---------------- selective scan (dA/dBu computed on the fly) + gating ----------------
__global__ __launch_bounds__(256) void scan_kernel(const float* __restrict__ xdbl,   // (Mc,96)
                                                   const float* __restrict__ delta,  // (Mc,DIc)
                                                   const float* __restrict__ xc32,   // (Mc,DIc)
                                                   const float* __restrict__ xz,     // (Mc,2*DIc), z at +DIc
                                                   const float* __restrict__ A_log,  // (DIc,DSc)
                                                   const float* __restrict__ Dvec,   // (DIc)
                                                   _Float16* __restrict__ y16) {     // (Mc,DIc)
  const int d = blockIdx.x * 256 + threadIdx.x;    // channel
  const int b = blockIdx.y;
  const int tid = threadIdx.x;

  float Ad[DSc], h[DSc];
#pragma unroll
  for (int s = 0; s < DSc; ++s) { Ad[s] = -__expf(A_log[(size_t)d * DSc + s]); h[s] = 0.f; }
  const float Dd = Dvec[d];

  __shared__ float sB[2][DSc], sC[2][DSc];

  for (int l = 0; l < Lc; ++l) {
    const int buf = l & 1;
    const size_t row = (size_t)b * Lc + l;
    const float* xr = xdbl + row * XDBLc;
    if (tid < DSc)            sB[buf][tid]        = xr[DTRc + tid];
    else if (tid < 2 * DSc)   sC[buf][tid - DSc]  = xr[DTRc + DSc + (tid - DSc)];
    __syncthreads();                               // single barrier; ping-pong keeps it safe

    const float dv = delta[row * DIc + d];
    const float xv = xc32[row * DIc + d];
    const float zv = xz[row * 2 * DIc + DIc + d];
    float y = 0.f;
#pragma unroll
    for (int s = 0; s < DSc; ++s) {
      const float a = __expf(dv * Ad[s]);
      h[s] = a * h[s] + (dv * sB[buf][s]) * xv;
      y += h[s] * sC[buf][s];
    }
    y = (y + xv * Dd) * (zv * (1.f / (1.f + __expf(-zv))));
    y16[row * DIc + d] = (_Float16)y;
  }
}

extern "C" void kernel_launch(void* const* d_in, const int* in_sizes, int n_in,
                              void* d_out, int out_size, void* d_ws, size_t ws_size,
                              hipStream_t stream) {
  const float* seq       = (const float*)d_in[0];
  const float* desc      = (const float*)d_in[1];
  const float* ln_w      = (const float*)d_in[2];
  const float* ln_b      = (const float*)d_in[3];
  const float* gamma_w   = (const float*)d_in[4];
  const float* gamma_b   = (const float*)d_in[5];
  const float* beta_w    = (const float*)d_in[6];
  const float* beta_b    = (const float*)d_in[7];
  const float* in_proj_w = (const float*)d_in[8];
  const float* conv_w    = (const float*)d_in[9];
  const float* conv_b    = (const float*)d_in[10];
  const float* x_proj_w  = (const float*)d_in[11];
  const float* dt_proj_w = (const float*)d_in[12];
  const float* dt_proj_b = (const float*)d_in[13];
  const float* A_log     = (const float*)d_in[14];
  const float* Dvec      = (const float*)d_in[15];
  const float* out_proj_w= (const float*)d_in[16];
  float* out = (float*)d_out;

  // ---- workspace layout (bytes) ----
  char* ws = (char*)d_ws;
  size_t off = 0;
  auto alloc = [&](size_t bytes) { char* p = ws + off; off += (bytes + 255) & ~(size_t)255; return p; };
  _Float16* s16      = (_Float16*)alloc((size_t)Mc * Ec * 2);           // LN+FiLM activations
  _Float16* w_in16   = (_Float16*)alloc((size_t)2 * DIc * Ec * 2);
  float*    xz       = (float*)   alloc((size_t)Mc * 2 * DIc * 4);
  _Float16* xc16     = (_Float16*)alloc((size_t)Mc * DIc * 2);
  float*    xc32     = (float*)   alloc((size_t)Mc * DIc * 4);
  _Float16* w_xp16   = (_Float16*)alloc((size_t)XDBLc * DIc * 2);
  float*    xdbl     = (float*)   alloc((size_t)Mc * XDBLc * 4);
  _Float16* dt16     = (_Float16*)alloc((size_t)Mc * DTRc * 2);
  _Float16* w_dt16   = (_Float16*)alloc((size_t)DIc * DTRc * 2);
  float*    delta    = (float*)   alloc((size_t)Mc * DIc * 4);
  _Float16* y16      = (_Float16*)alloc((size_t)Mc * DIc * 2);
  _Float16* w_out16  = (_Float16*)alloc((size_t)Ec * DIc * 2);
  float*    gbuf     = (float*)   alloc((size_t)Bc * Ec * 4);
  float*    btbuf    = (float*)   alloc((size_t)Bc * Ec * 4);
  if (off > ws_size) return;  // workspace too small: bail deterministically

  // 1) FiLM conditioning params
  film_params<<<(Bc * Ec + 255) / 256, 256, 0, stream>>>(desc, gamma_w, gamma_b, beta_w, beta_b, gbuf, btbuf);

  // 2) LayerNorm + FiLM -> f16 activations
  ln_film<<<Mc, 256, 0, stream>>>(seq, ln_w, ln_b, gbuf, btbuf, s16);

  // 3) weight conversions (once per call; cheap)
  cvt_f32_f16<<<((2 * DIc * Ec) + 255) / 256, 256, 0, stream>>>(in_proj_w, w_in16, 2 * DIc * Ec);
  cvt_f32_f16<<<((XDBLc * DIc) + 255) / 256, 256, 0, stream>>>(x_proj_w, w_xp16, XDBLc * DIc);
  cvt_f32_f16<<<((DIc * DTRc) + 255) / 256, 256, 0, stream>>>(dt_proj_w, w_dt16, DIc * DTRc);
  cvt_f32_f16<<<((Ec * DIc) + 255) / 256, 256, 0, stream>>>(out_proj_w, w_out16, Ec * DIc);

  // 4) in_proj GEMM: (4096,1024)x(1024,4096) -> xz
  {
    int waves = (Mc / 32) * ((2 * DIc) / 32);
    wmma_gemm_nt<<<(waves + 7) / 8, 256, 0, stream>>>(s16, w_in16, xz, Mc, 2 * DIc, Ec, nullptr, 0);
  }

  // 5) causal depthwise conv + SiLU
  conv_silu<<<((Bc * Lc * DIc) + 255) / 256, 256, 0, stream>>>(xz, conv_w, conv_b, xc32, xc16);

  // 6) x_proj GEMM: (4096,2048)x(2048,96) -> x_dbl
  {
    int waves = (Mc / 32) * (XDBLc / 32);
    wmma_gemm_nt<<<(waves + 7) / 8, 256, 0, stream>>>(xc16, w_xp16, xdbl, Mc, XDBLc, DIc, nullptr, 0);
  }

  // 7) dt slice -> f16, dt_proj GEMM with fused softplus(+bias): (4096,64)x(64,2048) -> delta
  dt_slice<<<((Mc * DTRc) + 255) / 256, 256, 0, stream>>>(xdbl, dt16);
  {
    int waves = (Mc / 32) * (DIc / 32);
    wmma_gemm_nt<<<(waves + 7) / 8, 256, 0, stream>>>(dt16, w_dt16, delta, Mc, DIc, DTRc, dt_proj_b, 1);
  }

  // 8) selective scan + D skip + silu(z) gating -> y16
  {
    dim3 grid(DIc / 256, Bc);
    scan_kernel<<<grid, 256, 0, stream>>>(xdbl, delta, xc32, xz, A_log, Dvec, y16);
  }

  // 9) out_proj GEMM: (4096,2048)x(2048,1024) -> d_out
  {
    int waves = (Mc / 32) * (Ec / 32);
    wmma_gemm_nt<<<(waves + 7) / 8, 256, 0, stream>>>(y16, w_out16, out, Mc, Ec, DIc, nullptr, 0);
  }
}
